// Attention_73547019976944
// MI455X (gfx1250) — matrix operand order repaired and checked by reading, not compiled, not run on hardware
//
#include <hip/hip_runtime.h>
#include <hip/hip_bf16.h>

#define N_HEADS 16
#define N_KV_HEADS 4
#define HEAD_DIM 128
#define DIM 2048
#define SEQ 2048
#define BATCH 2

typedef __attribute__((ext_vector_type(16))) __bf16 v16bf;
typedef __attribute__((ext_vector_type(8)))  __bf16 v8bf;
typedef __attribute__((ext_vector_type(8)))  float  v8f;
typedef int v4i128 __attribute__((__vector_size__(16)));   // matches builtin param

// ---- CDNA5 async-to-LDS feature detection (fallback = sync loads) ----------
#if defined(__has_builtin)
#if __has_builtin(__builtin_amdgcn_global_load_async_to_lds_b128)
#define HAVE_ASYNC_LDS 1
#endif
#if __has_builtin(__builtin_amdgcn_s_wait_asynccnt)
#define HAVE_WAIT_ASYNC_BUILTIN 1
#endif
#endif

#define AS1 __attribute__((address_space(1)))
#define AS3 __attribute__((address_space(3)))

// LDS offset is the low 32 bits of the generic address (aperture rule);
// global flat address == global address. Integer casts always compile.
__device__ inline AS3 v4i128* lds_cast4(void* p) {
  return (AS3 v4i128*)(unsigned)(unsigned long long)p;
}
__device__ inline AS1 v4i128* gbl_cast4(const void* p) {
  return (AS1 v4i128*)(unsigned long long)p;
}

__device__ inline void wait_async0() {
#if defined(HAVE_WAIT_ASYNC_BUILTIN)
  __builtin_amdgcn_s_wait_asynccnt(0);
#else
  asm volatile("s_wait_asynccnt 0x0" ::: "memory");
#endif
}

__device__ inline v8f wmma_bf16(v16bf a, v16bf b, v8f c) {
  // D = A(16x32 bf16) * B(32x16 bf16) + C(16x16 f32)
  return __builtin_amdgcn_wmma_f32_16x16x32_bf16(
      /*neg_a=*/false, a, /*neg_b=*/false, b,
      /*c_mod=*/(short)0, c, /*reuse_a=*/false, /*reuse_b=*/false);
}

// Build an A-fragment (16x32, contiguous-K layout, k = 16h + s) from two aligned 8-elem runs.
__device__ inline v16bf mk_afrag(const __bf16* lo_p, const __bf16* hi_p) {
  v8bf lo = *(const v8bf*)lo_p;
  v8bf hi = *(const v8bf*)hi_p;
  v16bf r;
#pragma unroll
  for (int e = 0; e < 8; ++e) { r[e] = lo[e]; r[8 + e] = hi[e]; }
  return r;
}

// 8-element load + convert-to-bf16 into LDS (contiguous dst).
__device__ inline void ldcvt8(__bf16* dst, const float* src) {
  const float4* p = (const float4*)src;
  float4 a = p[0], b = p[1];
  dst[0] = (__bf16)a.x; dst[1] = (__bf16)a.y; dst[2] = (__bf16)a.z; dst[3] = (__bf16)a.w;
  dst[4] = (__bf16)b.x; dst[5] = (__bf16)b.y; dst[6] = (__bf16)b.z; dst[7] = (__bf16)b.w;
}
__device__ inline void ldcvt8(__bf16* dst, const __bf16* src) {
  *(v8bf*)dst = *(const v8bf*)src;
}

__device__ inline void stc(float* p, float v)  { *p = v; }
__device__ inline void stc(__bf16* p, float v) { *p = (__bf16)v; }

// ---------------------------------------------------------------------------
// Generic bf16 WMMA GEMM: C[MxN] = A[MxK] * B[KxN], B is always f32 (weights),
// A is f32 (activations) or bf16 (attention output), C is bf16 or f32.
// Block tile 128x128, 4 waves, 64x64 per wave, K-step 32.
// bf16 A tiles are staged with GLOBAL_LOAD_ASYNC_TO_LDS when available.
// ---------------------------------------------------------------------------
template <typename TA, typename TC>
__global__ __launch_bounds__(128, 1) void gemm_wmma(const TA* __restrict__ A,
                                                    const float* __restrict__ Bm,
                                                    TC* __restrict__ C,
                                                    int M, int N, int K) {
  __shared__ __align__(32) __bf16 Alds[128][32];   // [m][k]
  __shared__ __align__(32) __bf16 Blds[128][32];   // transposed: [n][k]

  const int tid  = threadIdx.x;
  const int lane = tid & 31;
  const int wave = tid >> 5;
  const int m0 = blockIdx.y * 128;
  const int n0 = blockIdx.x * 128;
  const int wm = (wave >> 1) * 64;
  const int wn = (wave & 1) * 64;
  const int row16 = lane & 15;
  const int kh   = (lane & 16) ? 8 : 0;    // unused; A-frag uses kb16
  const int kb16 = (lane & 16) ? 16 : 0;   // B-frag half offset
  const int mhi  = (lane & 16) ? 8 : 0;    // C row offset

  v8f acc[4][4];
#pragma unroll
  for (int i = 0; i < 4; ++i)
#pragma unroll
    for (int j = 0; j < 4; ++j) acc[i][j] = (v8f){0.f,0.f,0.f,0.f,0.f,0.f,0.f,0.f};

  for (int k0 = 0; k0 < K; k0 += 32) {
    __syncthreads();
    // --- A tile: 128x32, 8-elem chunks, coalesced; bf16 goes async->LDS ---
#pragma unroll
    for (int i = 0; i < 4; ++i) {
      int idx = tid + i * 128;              // 0..511
      int r = idx >> 2;                     // 0..127
      int c = (idx & 3) * 8;                // 0,8,16,24
      const TA* src = A + (size_t)(m0 + r) * K + (k0 + c);
#if defined(HAVE_ASYNC_LDS)
      if constexpr (sizeof(TA) == 2) {
        __builtin_amdgcn_global_load_async_to_lds_b128(
            gbl_cast4(src), lds_cast4(&Alds[r][c]), 0, 0);
      } else
#endif
      {
        ldcvt8(&Alds[r][c], src);
      }
    }
    // --- B tile: 32x128 read coalesced, stored transposed [n][k] ---
#pragma unroll
    for (int i = 0; i < 4; ++i) {
      int idx = tid + i * 128;              // 0..511
      int r = idx >> 4;                     // k: 0..31
      int c = (idx & 15) * 8;               // n: 0..120
      const float4* p = (const float4*)(Bm + (size_t)(k0 + r) * N + n0 + c);
      float4 u = p[0], v = p[1];
      Blds[c + 0][r] = (__bf16)u.x; Blds[c + 1][r] = (__bf16)u.y;
      Blds[c + 2][r] = (__bf16)u.z; Blds[c + 3][r] = (__bf16)u.w;
      Blds[c + 4][r] = (__bf16)v.x; Blds[c + 5][r] = (__bf16)v.y;
      Blds[c + 6][r] = (__bf16)v.z; Blds[c + 7][r] = (__bf16)v.w;
    }
#if defined(HAVE_ASYNC_LDS)
    if constexpr (sizeof(TA) == 2) wait_async0();
#endif
    __syncthreads();

    v16bf afrag[4], bfrag[4];
#pragma unroll
    for (int f = 0; f < 4; ++f) {
      int ar = wm + f * 16 + row16;
      afrag[f] = mk_afrag(&Alds[ar][kb16], &Alds[ar][kb16 + 8]);
      bfrag[f] = *(const v16bf*)&Blds[wn + f * 16 + row16][kb16];
    }
#pragma unroll
    for (int i = 0; i < 4; ++i)
#pragma unroll
      for (int j = 0; j < 4; ++j)
        acc[i][j] = wmma_bf16(afrag[i], bfrag[j], acc[i][j]);
  }

  // Epilogue: C/D layout — VGPR r, lanes<16: M=r, lanes>=16: M=8+r; N=lane%16.
#pragma unroll
  for (int i = 0; i < 4; ++i)
#pragma unroll
    for (int j = 0; j < 4; ++j)
#pragma unroll
      for (int r = 0; r < 8; ++r) {
        int m = m0 + wm + i * 16 + mhi + r;
        int n = n0 + wn + j * 16 + row16;
        stc(&C[(size_t)m * N + n], acc[i][j][r]);
      }
}

// ---------------------------------------------------------------------------
// Interleaved RoPE on bf16 tensor laid out [(b*S+s)*heads + h][128].
// ---------------------------------------------------------------------------
__global__ void rope_kernel(__bf16* __restrict__ t, const float* __restrict__ fc,
                            const float* __restrict__ fs, int heads, int total) {
  int idx = blockIdx.x * blockDim.x + threadIdx.x;
  if (idx >= total) return;
  int i   = idx & 63;
  int row = idx >> 6;                 // (b*S+s)*heads + h
  int s   = (row / heads) % SEQ;
  float c  = fc[s * 64 + i];
  float sn = fs[s * 64 + i];
  size_t base = (size_t)row * HEAD_DIM + 2 * i;
  float te = (float)t[base];
  float to = (float)t[base + 1];
  t[base]     = (__bf16)(te * c - to * sn);
  t[base + 1] = (__bf16)(te * sn + to * c);
}

// ---------------------------------------------------------------------------
// Flash attention, causal, GQA (16 Q heads, 4 KV heads).
// Grid: (SEQ/64, BATCH*N_HEADS). Block: 128 threads (4 waves).
// K tiles staged via async-to-LDS; V transposed in flight (sync path);
// next tile prefetched with global_prefetch_b8.
// ---------------------------------------------------------------------------
__global__ __launch_bounds__(128, 1) void flash_attn(const __bf16* __restrict__ Q,
                                                     const __bf16* __restrict__ Kt,
                                                     const __bf16* __restrict__ Vt,
                                                     __bf16* __restrict__ O) {
  __shared__ __align__(32) __bf16 Klds[32][128];     // [key][feat]
  __shared__ __align__(32) __bf16 Vlds[128][32];     // transposed: [feat][key]
  __shared__ __align__(32) __bf16 Plds[4][16][32];   // per-wave P tile [qrow][key]

  const int tid  = threadIdx.x;
  const int lane = tid & 31;
  const int wave = tid >> 5;
  const int qblk = blockIdx.x;
  const int bh   = blockIdx.y;
  const int b    = bh >> 4;
  const int h    = bh & 15;
  const int kvh  = h >> 2;
  const int row16 = lane & 15;
  const int kh    = (lane & 16) ? 8 : 0;
  const int kb16  = (lane & 16) ? 16 : 0;
  const int mhi   = (lane & 16) ? 8 : 0;
  const int qbase = qblk * 64 + wave * 16;
  const float scale = 0.08838834764831845f;   // 1/sqrt(128)

  // Q fragments for this wave's 16 rows (layout [(b*S+s)*16+h][128])
  v16bf qf[4];
  {
    const __bf16* qrow = Q + ((size_t)((b * SEQ + qbase + row16) * N_HEADS + h)) * HEAD_DIM;
#pragma unroll
    for (int c = 0; c < 4; ++c)
      qf[c] = mk_afrag(qrow + c * 32 + kb16, qrow + c * 32 + kb16 + 8);
  }

  v8f out[8];
#pragma unroll
  for (int t = 0; t < 8; ++t) out[t] = (v8f){0.f,0.f,0.f,0.f,0.f,0.f,0.f,0.f};
  float mrow[8], lrow[8];
#pragma unroll
  for (int r = 0; r < 8; ++r) { mrow[r] = -3.0e38f; lrow[r] = 0.f; }

  const int nkb = qblk * 2 + 2;   // key blocks of 32 covering keys <= qblk*64+63
  for (int kb = 0; kb < nkb; ++kb) {
    __syncthreads();
    // Cooperative K/V tile load (32 keys x 128 feats).
#pragma unroll
    for (int i = 0; i < 4; ++i) {
      int idx = tid + i * 128;                // 0..511
      int r = idx >> 4;                       // key 0..31
      int c = (idx & 15) * 8;                 // feat 0..120
      size_t g = ((size_t)((b * SEQ + kb * 32 + r) * N_KV_HEADS + kvh)) * HEAD_DIM + c;
#if defined(HAVE_ASYNC_LDS)
      __builtin_amdgcn_global_load_async_to_lds_b128(
          gbl_cast4(Kt + g), lds_cast4(&Klds[r][c]), 0, 0);
#else
      *(v8bf*)&Klds[r][c] = *(const v8bf*)(Kt + g);
#endif
      v8bf vv = *(const v8bf*)(Vt + g);
#pragma unroll
      for (int j = 0; j < 8; ++j) Vlds[c + j][r] = vv[j];
      if (kb + 1 < nkb) {   // prefetch next key block (32 keys ahead)
        __builtin_prefetch(Kt + g + 32 * N_KV_HEADS * HEAD_DIM, 0, 1);
        __builtin_prefetch(Vt + g + 32 * N_KV_HEADS * HEAD_DIM, 0, 1);
      }
    }
#if defined(HAVE_ASYNC_LDS)
    wait_async0();
#endif
    __syncthreads();

    // Scores: S(16q x 32k) = Q(16x128) * K^T, two 16x16 WMMA accumulators.
    v8f s0 = (v8f){0.f,0.f,0.f,0.f,0.f,0.f,0.f,0.f};
    v8f s1 = s0;
#pragma unroll
    for (int c = 0; c < 4; ++c) {
      v16bf b0 = *(const v16bf*)&Klds[row16][c * 32 + kb16];        // keys 0..15
      v16bf b1 = *(const v16bf*)&Klds[16 + row16][c * 32 + kb16];   // keys 16..31
      s0 = wmma_bf16(qf[c], b0, s0);
      s1 = wmma_bf16(qf[c], b1, s1);
    }

    // Online softmax update (rows live in 16-lane halves; reduce with shfl_xor).
#pragma unroll
    for (int r = 0; r < 8; ++r) {
      int qg   = qbase + mhi + r;
      int key0 = kb * 32 + row16;
      float a0 = (key0      <= qg) ? s0[r] * scale : -3.0e38f;
      float a1 = (key0 + 16 <= qg) ? s1[r] * scale : -3.0e38f;
      float mx = fmaxf(a0, a1);
#pragma unroll
      for (int off = 8; off >= 1; off >>= 1) mx = fmaxf(mx, __shfl_xor(mx, off, 32));
      float mn   = fmaxf(mrow[r], mx);
      float corr = __expf(mrow[r] - mn);
      float p0   = __expf(a0 - mn);
      float p1   = __expf(a1 - mn);
      float sum  = p0 + p1;
#pragma unroll
      for (int off = 8; off >= 1; off >>= 1) sum += __shfl_xor(sum, off, 32);
      lrow[r] = lrow[r] * corr + sum;
      mrow[r] = mn;
#pragma unroll
      for (int t = 0; t < 8; ++t) out[t][r] *= corr;
      Plds[wave][mhi + r][row16]      = (__bf16)p0;
      Plds[wave][mhi + r][16 + row16] = (__bf16)p1;
    }

    // Wave-internal LDS visibility for the C->A layout round-trip of P.
    asm volatile("s_wait_dscnt 0" ::: "memory");
    __builtin_amdgcn_wave_barrier();

    // O(16x128) += P(16x32) * V(32x128)
    v16bf pf = mk_afrag(&Plds[wave][row16][kb16], &Plds[wave][row16][kb16 + 8]);
#pragma unroll
    for (int t = 0; t < 8; ++t) {
      v16bf vf = *(const v16bf*)&Vlds[t * 16 + row16][kb16];
      out[t] = wmma_bf16(pf, vf, out[t]);
    }
  }

  // Normalize and scatter to attention buffer [(b*S+q)*16+h][128] (bf16).
#pragma unroll
  for (int r = 0; r < 8; ++r) {
    float inv = 1.0f / lrow[r];
    int qg = qbase + mhi + r;
    size_t ob = ((size_t)((b * SEQ + qg) * N_HEADS + h)) * HEAD_DIM;
#pragma unroll
    for (int t = 0; t < 8; ++t)
      O[ob + t * 16 + row16] = (__bf16)(out[t][r] * inv);
  }
}

// ---------------------------------------------------------------------------
extern "C" void kernel_launch(void* const* d_in, const int* in_sizes, int n_in,
                              void* d_out, int out_size, void* d_ws, size_t ws_size,
                              hipStream_t stream) {
  const float* x  = (const float*)d_in[0];   // [B,S,DIM] f32
  const float* wq = (const float*)d_in[1];   // [DIM, 2048]
  const float* wk = (const float*)d_in[2];   // [DIM, 512]
  const float* wv = (const float*)d_in[3];   // [DIM, 512]
  const float* wo = (const float*)d_in[4];   // [2048, DIM]
  const float* fc = (const float*)d_in[5];   // [SEQ, 64]
  const float* fs = (const float*)d_in[6];   // [SEQ, 64]
  // d_in[7] = mask, unused (causality computed in-kernel)
  float* out = (float*)d_out;

  const int M = BATCH * SEQ;                  // 4096
  __bf16* Qb = (__bf16*)d_ws;                 // 4096*2048 = 8M elems (16 MB)
  __bf16* Kb = Qb + (size_t)M * 2048;         // 4096*512  = 2M elems (4 MB)
  __bf16* Vb = Kb + (size_t)M * 512;          // 2M elems (4 MB)
  __bf16* Ab = Vb + (size_t)M * 512;          // 8M elems (16 MB); total 40 MB

  dim3 blk(128);
  // Projections (fused f32->bf16 conversion inside the GEMM tile loaders)
  gemm_wmma<float, __bf16><<<dim3(2048 / 128, M / 128), blk, 0, stream>>>(x, wq, Qb, M, 2048, DIM);
  gemm_wmma<float, __bf16><<<dim3(512  / 128, M / 128), blk, 0, stream>>>(x, wk, Kb, M, 512,  DIM);
  gemm_wmma<float, __bf16><<<dim3(512  / 128, M / 128), blk, 0, stream>>>(x, wv, Vb, M, 512,  DIM);
  // RoPE on Q (16 heads) and K (4 KV heads)
  rope_kernel<<<(M * N_HEADS * 64)    / 256, 256, 0, stream>>>(Qb, fc, fs, N_HEADS,    M * N_HEADS * 64);
  rope_kernel<<<(M * N_KV_HEADS * 64) / 256, 256, 0, stream>>>(Kb, fc, fs, N_KV_HEADS, M * N_KV_HEADS * 64);
  // Causal flash attention
  flash_attn<<<dim3(SEQ / 64, BATCH * N_HEADS), blk, 0, stream>>>(Qb, Kb, Vb, Ab);
  // Output projection (bf16 A, f32 weights, f32 out)
  gemm_wmma<__bf16, float><<<dim3(16, 32), blk, 0, stream>>>(Ab, wo, out, M, DIM, 2048);
}
